// Attention_52166672778096
// MI455X (gfx1250) — matrix operand order repaired and checked by reading, not compiled, not run on hardware
//
#include <hip/hip_runtime.h>

// ---------------------------------------------------------------------------
// Types for CDNA5 WMMA (wave32, 16x16x32 bf16 -> f32)
// ---------------------------------------------------------------------------
typedef __attribute__((ext_vector_type(16))) __bf16 v16bf;
typedef __attribute__((ext_vector_type(8)))  __bf16 v8bf;
typedef __attribute__((ext_vector_type(8)))  float  v8f;
typedef __attribute__((ext_vector_type(8)))  unsigned short v8us;

union V16U { v16bf v; v8bf h[2]; };

// Problem constants
#define BB   2
#define SS   2048
#define DD   1024
#define HH   16
#define HD   64
#define MTOT (BB * SS)          // 4096
#define N3D  (3 * DD)           // 3072

// log2(e) folded into the q scale so softmax uses exp2 directly
#define QSCALE (0.125f * 1.44269504088896340736f)

// Workspace layout (bytes)
#define OFF_XBF  ((size_t)0)                         // [4096,1024] bf16   8.39MB
#define OFF_WABF ((size_t)8388608)                   // Wt: [3072,1024] bf16 (transposed)
#define OFF_WPBF ((size_t)14680064)                  // Wt: [1024,1024] bf16 (transposed)
#define OFF_Q    ((size_t)16777216)                  // [B,H,S,64]  bf16
#define OFF_K    ((size_t)25165824)                  // [B,H,S,64]  bf16
#define OFF_VT   ((size_t)33554432)                  // [B,H,64,S]  bf16
#define OFF_Z    ((size_t)41943040)                  // [B,S,1024]  bf16
// total ~50.3MB

__device__ __forceinline__ unsigned short f2bf(float f) {
  unsigned int u = __float_as_uint(f);
  u += 0x7fffu + ((u >> 16) & 1u);      // round to nearest even
  return (unsigned short)(u >> 16);
}

// A fragment: 16x32 (MxK), row-major source, 16-bit elements.
// Lane l: row = l&15; K chunks at [g*8, +8) and [16+g*8, +8), g = l>>4.
__device__ __forceinline__ v16bf load_a_frag(const unsigned short* base, int ldk, int lane) {
  int row = lane & 15;
  int g   = lane >> 4;
  const unsigned short* p = base + row * ldk + g * 8;
  V16U u;
  u.h[0] = *(const v8bf*)(p);
  u.h[1] = *(const v8bf*)(p + 16);
  return u.v;
}

// B fragment: 32x16 (KxN), source laid out [n][k] (n-major, K contiguous).
// Lane l: n = l&15; K chunks at [g*8, +8) and [16+g*8, +8), g = l>>4.
__device__ __forceinline__ v16bf load_b_frag(const unsigned short* base, int ldk, int lane) {
  int n = lane & 15;
  int g = lane >> 4;
  const unsigned short* p = base + n * ldk + g * 8;
  V16U u;
  u.h[0] = *(const v8bf*)(p);
  u.h[1] = *(const v8bf*)(p + 16);
  return u.v;
}

__device__ __forceinline__ v8f wmma_bf16(v16bf a, v16bf b, v8f c) {
  return __builtin_amdgcn_wmma_f32_16x16x32_bf16(false, a, false, b, (short)0, c, false, false);
}

// ---------------------------------------------------------------------------
// fp32 -> bf16 (vectorized: float2 in, packed 2xbf16 u32 out)
// ---------------------------------------------------------------------------
__global__ __launch_bounds__(256) void cvt_bf16_kernel(const float2* __restrict__ src,
                                                       unsigned int* __restrict__ dst, int n2) {
  int i = blockIdx.x * 256 + threadIdx.x;
  int stride = gridDim.x * 256;
  for (; i < n2; i += stride) {
    float2 f = src[i];
    dst[i] = ((unsigned int)f2bf(f.y) << 16) | (unsigned int)f2bf(f.x);
  }
}

// ---------------------------------------------------------------------------
// fp32 [K,N] -> bf16 transposed [N,K]  (64x64 tile through LDS; coalesced
// float reads, b128 LDS reads, 32B global stores). K is always DD (=1024).
// ---------------------------------------------------------------------------
__global__ __launch_bounds__(256) void cvt_t_kernel(const float* __restrict__ W,
                                                    unsigned short* __restrict__ Wt, int Ncols) {
  __shared__ unsigned short T[64][72];     // pad 8: row stride 144B (16B aligned)
  const int nTiles = Ncols >> 6;
  const int k0 = (blockIdx.x / nTiles) << 6;
  const int n0 = (blockIdx.x % nTiles) << 6;
  const int tid = threadIdx.x;
#pragma unroll
  for (int i = tid; i < 64 * 64; i += 256) {
    int kk = i >> 6;
    int nn = i & 63;                       // consecutive tid -> consecutive n: coalesced
    T[nn][kk] = f2bf(W[(size_t)(k0 + kk) * Ncols + n0 + nn]);
  }
  __syncthreads();
  const int nn = tid >> 2;
  const int kc = (tid & 3) << 4;
  v8us* dst = (v8us*)(Wt + (size_t)(n0 + nn) * DD + k0 + kc);
  dst[0] = *(const v8us*)(&T[nn][kc]);
  dst[1] = *(const v8us*)(&T[nn][kc + 8]);
}

// ---------------------------------------------------------------------------
// Barrier-free GEMM core: each wave computes a 32(M) x 64(N) tile.
// All 6 fragments of a K-step load into distinct registers first (one load
// clause, overlapped waits), then 8 WMMAs consume them.
// ---------------------------------------------------------------------------
__device__ __forceinline__ void gemm_core(const unsigned short* __restrict__ X,
                                          const unsigned short* __restrict__ Wt,
                                          int bm, int bn, int wave, int lane,
                                          v8f acc[2][4]) {
  const int rowBase = bm + wave * 32;
  const unsigned short* Arow = X + (size_t)rowBase * DD;
  const unsigned short* Brow = Wt + (size_t)bn * DD;
#pragma unroll 2
  for (int k0 = 0; k0 < DD; k0 += 32) {
    if (k0 + 32 < DD) {
      __builtin_prefetch(Arow + k0 + 32, 0, 3);                 // near-scope prefetch
      __builtin_prefetch(Brow + k0 + 32, 0, 3);
    }
    v16bf a0 = load_a_frag(Arow + k0,           DD, lane);
    v16bf a1 = load_a_frag(Arow + 16 * DD + k0, DD, lane);
    v16bf b0 = load_b_frag(Brow +  0 * DD + k0, DD, lane);
    v16bf b1 = load_b_frag(Brow + 16 * DD + k0, DD, lane);
    v16bf b2 = load_b_frag(Brow + 32 * DD + k0, DD, lane);
    v16bf b3 = load_b_frag(Brow + 48 * DD + k0, DD, lane);
    acc[0][0] = wmma_bf16(a0, b0, acc[0][0]);
    acc[1][0] = wmma_bf16(a1, b0, acc[1][0]);
    acc[0][1] = wmma_bf16(a0, b1, acc[0][1]);
    acc[1][1] = wmma_bf16(a1, b1, acc[1][1]);
    acc[0][2] = wmma_bf16(a0, b2, acc[0][2]);
    acc[1][2] = wmma_bf16(a1, b2, acc[1][2]);
    acc[0][3] = wmma_bf16(a0, b3, acc[0][3]);
    acc[1][3] = wmma_bf16(a1, b3, acc[1][3]);
  }
}

// ---------------------------------------------------------------------------
// QKV GEMM: hidden = x @ w_attn + b_attn; scatter into Q (scaled), K, V^T (bf16)
// ---------------------------------------------------------------------------
__global__ __launch_bounds__(256) void gemm_qkv_kernel(const unsigned short* __restrict__ X,
                                                       const unsigned short* __restrict__ Wt,
                                                       const float* __restrict__ bias,
                                                       unsigned short* __restrict__ Qb,
                                                       unsigned short* __restrict__ Kb,
                                                       unsigned short* __restrict__ Vt) {
  const int bid  = blockIdx.x;
  const int bm   = (bid / (N3D / 64)) * 256;
  const int bn   = (bid % (N3D / 64)) * 64;
  const int wave = threadIdx.x >> 5;
  const int lane = threadIdx.x & 31;

  v8f acc[2][4] = {};
  gemm_core(X, Wt, bm, bn, wave, lane, acc);

  const int hgrp = lane >> 4;
  const int nl   = lane & 15;
#pragma unroll
  for (int g = 0; g < 4; ++g) {
    int col = bn + g * 16 + nl;
    float bv = bias[col];
    int which = col >> 10;          // 0=q 1=k 2=v
    int dd    = col & 1023;
    int hh    = dd >> 6;
    int hd    = dd & 63;
#pragma unroll
    for (int mi = 0; mi < 2; ++mi) {
#pragma unroll
      for (int r = 0; r < 8; ++r) {
        int row = bm + wave * 32 + mi * 16 + hgrp * 8 + r;   // global m = b*S + s
        int b = row >> 11;
        int s = row & (SS - 1);
        float v = acc[mi][g][r] + bv;
        if (which == 0) {
          Qb[(((b * HH + hh) * SS) + s) * HD + hd] = f2bf(v * QSCALE);
        } else if (which == 1) {
          Kb[(((b * HH + hh) * SS) + s) * HD + hd] = f2bf(v);
        } else {
          Vt[(((b * HH + hh) * HD) + hd) * SS + s] = f2bf(v);   // transposed
        }
      }
    }
  }
}

// ---------------------------------------------------------------------------
// Flash attention: one wave owns 16 q rows; loop 32-key tiles with online
// softmax. K and V^T fragments load straight from global (layouts match the
// WMMA B-fragment pattern); V-frags are issued BEFORE the softmax so their
// latency hides under the VALU work. P transposes C->A layout through LDS.
// ---------------------------------------------------------------------------
__global__ __launch_bounds__(256) void attn_kernel(const unsigned short* __restrict__ Q,
                                                   const unsigned short* __restrict__ K,
                                                   const unsigned short* __restrict__ V,
                                                   unsigned short* __restrict__ Z) {
  __shared__ unsigned short P[8][16][40];   // per-wave 16x32 P tile (stride 40)
  const int bid  = blockIdx.x;              // B*H*(S/128) = 512
  const int qb   = (bid & 15) * 128;
  const int bh   = bid >> 4;                // 0..31
  const int wave = threadIdx.x >> 5;
  const int lane = threadIdx.x & 31;
  const int hgrp = lane >> 4;
  const int nl   = lane & 15;

  const unsigned short* Qp = Q + (size_t)bh * SS * HD;
  const unsigned short* Kp = K + (size_t)bh * SS * HD;
  const unsigned short* Vp = V + (size_t)bh * HD * SS;

  const int qw = qb + wave * 16;            // this wave's 16 q rows
  unsigned short (*Pw)[40] = P[wave];

  // Q A-fragments (hd 0-31, 32-63) stay in registers for the whole loop
  v16bf aq0 = load_a_frag(Qp + qw * HD,      HD, lane);
  v16bf aq1 = load_a_frag(Qp + qw * HD + 32, HD, lane);

  v8f zacc[4] = {};
  float m_i[8], l_i[8];
#pragma unroll
  for (int r = 0; r < 8; ++r) { m_i[r] = -3.0e38f; l_i[r] = 0.0f; }

  const int rowMax = qw + 15;
  for (int kb = 0; kb <= rowMax; kb += 32) {
    if (kb + 32 <= rowMax) {                        // prefetch next key tile
      __builtin_prefetch(Kp + (kb + 32) * HD, 0, 3);
      __builtin_prefetch(Vp + kb + 32, 0, 3);
    }
    // ---- scores: S = Q @ K^T (base-2 scaled), two 16-col tiles ----
    v16bf b00 = load_b_frag(Kp + (kb +  0) * HD,      HD, lane);
    v16bf b01 = load_b_frag(Kp + (kb +  0) * HD + 32, HD, lane);
    v16bf b10 = load_b_frag(Kp + (kb + 16) * HD,      HD, lane);
    v16bf b11 = load_b_frag(Kp + (kb + 16) * HD + 32, HD, lane);
    v8f s0 = {}, s1 = {};
    s0 = wmma_bf16(aq0, b00, s0);
    s0 = wmma_bf16(aq1, b01, s0);
    s1 = wmma_bf16(aq0, b10, s1);
    s1 = wmma_bf16(aq1, b11, s1);

    // V^T fragments: independent of softmax/P -> issue now, consume later
    v16bf bv0 = load_b_frag(Vp +  0 * SS + kb, SS, lane);
    v16bf bv1 = load_b_frag(Vp + 16 * SS + kb, SS, lane);
    v16bf bv2 = load_b_frag(Vp + 32 * SS + kb, SS, lane);
    v16bf bv3 = load_b_frag(Vp + 48 * SS + kb, SS, lane);

    // ---- causal mask: needed whenever max col (kb+31) can exceed min row (qw)
    if (kb + 31 > qw) {
#pragma unroll
      for (int r = 0; r < 8; ++r) {
        int row = qw + hgrp * 8 + r;
        if (kb + nl      > row) s0[r] = -3.0e38f;
        if (kb + 16 + nl > row) s1[r] = -3.0e38f;
      }
    }

    // ---- online softmax (per row; lanes 0-15 / 16-31 hold M=r / M=8+r) ----
#pragma unroll
    for (int r = 0; r < 8; ++r) {
      float mx = fmaxf(s0[r], s1[r]);
#pragma unroll
      for (int off = 8; off >= 1; off >>= 1) mx = fmaxf(mx, __shfl_xor(mx, off, 32));
      float mnew  = fmaxf(m_i[r], mx);
      float alpha = exp2f(m_i[r] - mnew);
      m_i[r] = mnew;
      float p0 = exp2f(s0[r] - mnew);
      float p1 = exp2f(s1[r] - mnew);
      float ps = p0 + p1;
#pragma unroll
      for (int off = 8; off >= 1; off >>= 1) ps += __shfl_xor(ps, off, 32);
      l_i[r] = l_i[r] * alpha + ps;
#pragma unroll
      for (int g = 0; g < 4; ++g) zacc[g][r] *= alpha;
      // P tile to LDS (C layout -> memory): row M = hgrp*8+r, cols nl, nl+16
      Pw[hgrp * 8 + r][nl]      = f2bf(p0);
      Pw[hgrp * 8 + r][16 + nl] = f2bf(p1);
    }

    // ---- Z += P @ V : A-frag reload of P from LDS ----
    v16bf ap = load_a_frag(&Pw[0][0], 40, lane);
    zacc[0] = wmma_bf16(ap, bv0, zacc[0]);
    zacc[1] = wmma_bf16(ap, bv1, zacc[1]);
    zacc[2] = wmma_bf16(ap, bv2, zacc[2]);
    zacc[3] = wmma_bf16(ap, bv3, zacc[3]);
  }

  // ---- epilogue: z /= l, write merged [B,S,D] in bf16 ----
  const int b  = bh >> 4;
  const int hh = bh & 15;
  unsigned short* Zp = Z + ((size_t)b * SS) * DD + hh * HD;
#pragma unroll
  for (int r = 0; r < 8; ++r) {
    float inv = 1.0f / l_i[r];
    int s = qw + hgrp * 8 + r;
#pragma unroll
    for (int g = 0; g < 4; ++g) {
      Zp[(size_t)s * DD + g * 16 + nl] = f2bf(zacc[g][r] * inv);
    }
  }
}

// ---------------------------------------------------------------------------
// Output projection: out = z @ w_proj + b_proj (fp32 out)
// ---------------------------------------------------------------------------
__global__ __launch_bounds__(256) void gemm_proj_kernel(const unsigned short* __restrict__ X,
                                                        const unsigned short* __restrict__ Wt,
                                                        const float* __restrict__ bias,
                                                        float* __restrict__ out) {
  const int bid  = blockIdx.x;
  const int bm   = (bid / (DD / 64)) * 256;
  const int bn   = (bid % (DD / 64)) * 64;
  const int wave = threadIdx.x >> 5;
  const int lane = threadIdx.x & 31;

  v8f acc[2][4] = {};
  gemm_core(X, Wt, bm, bn, wave, lane, acc);

  const int hgrp = lane >> 4;
  const int nl   = lane & 15;
#pragma unroll
  for (int g = 0; g < 4; ++g) {
    int col = bn + g * 16 + nl;
    float bv = bias[col];
#pragma unroll
    for (int mi = 0; mi < 2; ++mi) {
#pragma unroll
      for (int r = 0; r < 8; ++r) {
        int row = bm + wave * 32 + mi * 16 + hgrp * 8 + r;
        out[(size_t)row * DD + col] = acc[mi][g][r] + bv;
      }
    }
  }
}

// ---------------------------------------------------------------------------
extern "C" void kernel_launch(void* const* d_in, const int* in_sizes, int n_in,
                              void* d_out, int out_size, void* d_ws, size_t ws_size,
                              hipStream_t stream) {
  (void)in_sizes; (void)n_in; (void)out_size; (void)ws_size;
  const float* x      = (const float*)d_in[0];
  const float* w_attn = (const float*)d_in[1];
  const float* b_attn = (const float*)d_in[2];
  const float* w_proj = (const float*)d_in[3];
  const float* b_proj = (const float*)d_in[4];
  float* out = (float*)d_out;

  char* ws = (char*)d_ws;
  unsigned short* XBF  = (unsigned short*)(ws + OFF_XBF);
  unsigned short* WABF = (unsigned short*)(ws + OFF_WABF);   // transposed [N3D][DD]
  unsigned short* WPBF = (unsigned short*)(ws + OFF_WPBF);   // transposed [DD][DD]
  unsigned short* Qb   = (unsigned short*)(ws + OFF_Q);
  unsigned short* Kb   = (unsigned short*)(ws + OFF_K);
  unsigned short* Vt   = (unsigned short*)(ws + OFF_VT);
  unsigned short* Zb   = (unsigned short*)(ws + OFF_Z);

  cvt_bf16_kernel<<<2048, 256, 0, stream>>>((const float2*)x, (unsigned int*)XBF, (MTOT * DD) / 2);
  cvt_t_kernel<<<(DD / 64) * (N3D / 64), 256, 0, stream>>>(w_attn, WABF, N3D);
  cvt_t_kernel<<<(DD / 64) * (DD / 64), 256, 0, stream>>>(w_proj, WPBF, DD);

  gemm_qkv_kernel<<<(MTOT / 256) * (N3D / 64), 256, 0, stream>>>(XBF, WABF, b_attn, Qb, Kb, Vt);
  attn_kernel<<<BB * HH * (SS / 128), 256, 0, stream>>>(Qb, Kb, Vt, Zb);
  gemm_proj_kernel<<<(MTOT / 256) * (DD / 64), 256, 0, stream>>>(Zb, WPBF, b_proj, out);
}
